// Architecture_979252544349
// MI455X (gfx1250) — matrix operand order repaired and checked by reading, not compiled, not run on hardware
//
#include <hip/hip_runtime.h>

// ---------------------------------------------------------------------------
// Types for CDNA5 WMMA (wave32): v_wmma_f32_16x16x32_bf16
// ---------------------------------------------------------------------------
typedef __attribute__((ext_vector_type(16))) __bf16 v16bf;
typedef __attribute__((ext_vector_type(8)))  __bf16 v8bf;
typedef __attribute__((ext_vector_type(8)))  float  v8f;

__device__ __forceinline__ v8f wmma_bf16(v16bf a, v16bf b, v8f c) {
  return __builtin_amdgcn_wmma_f32_16x16x32_bf16(false, a, false, b, (short)0, c,
                                                 false, false);
}

// A-matrix fragment (16x32, MxK): lanes 0-15 row M=lane, K=0..15;
// lanes 16-31 same rows, K=16..31. Two 16B loads per lane.
__device__ __forceinline__ v16bf load_a_frag(const __bf16* base, int stride,
                                             int row, int kbase, int lane) {
  int r = row + (lane & 15);
  int kh = (lane >> 4) * 16;
  const __bf16* p = base + (size_t)r * stride + kbase + kh;
  v8bf lo = *(const v8bf*)p;
  v8bf hi = *(const v8bf*)(p + 8);
  v16bf o;
#pragma unroll
  for (int i = 0; i < 8; ++i) { o[i] = lo[i]; o[i + 8] = hi[i]; }
  return o;
}

// B-matrix fragment (32x16, KxN): lane holds column (lane&15), 16 contiguous
// K values starting at (lane>>4)*16, from [N,K] row-major storage.
__device__ __forceinline__ v16bf load_b_frag(const __bf16* base, int stride,
                                             int col, int kbase, int lane) {
  int c = col + (lane & 15);
  int ko = (lane >> 4) * 16;
  return *(const v16bf*)(base + (size_t)c * stride + kbase + ko);
}

// 16 contiguous bf16 from a 16B-aligned pointer (two 16B loads).
__device__ __forceinline__ v16bf load16_contig(const __bf16* p) {
  v8bf lo = *(const v8bf*)p;
  v8bf hi = *(const v8bf*)(p + 8);
  v16bf o;
#pragma unroll
  for (int i = 0; i < 8; ++i) { o[i] = lo[i]; o[i + 8] = hi[i]; }
  return o;
}

// CDNA5 async global->LDS copy (16B per lane, ASYNCcnt tracked).
__device__ __forceinline__ void async_copy_b128(unsigned lds_addr,
                                                const void* gaddr) {
  asm volatile("global_load_async_to_lds_b128 %0, %1, off"
               :: "v"(lds_addr), "v"((unsigned long long)(uintptr_t)gaddr)
               : "memory");
}
__device__ __forceinline__ void wait_async0() {
  asm volatile("s_wait_asynccnt 0" ::: "memory");
}

// ---------------------------------------------------------------------------
// f32 -> bf16 conversion
// ---------------------------------------------------------------------------
__global__ void cvt_bf16_kernel(const float* __restrict__ x,
                                __bf16* __restrict__ y, size_t n) {
  size_t i = (size_t)blockIdx.x * blockDim.x + threadIdx.x;
  size_t stride = (size_t)gridDim.x * blockDim.x;
  for (; i < n; i += stride) y[i] = (__bf16)x[i];
}

// ---------------------------------------------------------------------------
// GEMM: C[M,N] = A[M,K] @ W[N,K]^T + bias, optional ReLU. A,W bf16, C f32.
// Block = 256 threads = 8 waves; wave w owns 16(M)x64(N); block = 128x64.
// The 64x32 weight tile (shared by all 8 waves) is staged in LDS via
// double-buffered global_load_async_to_lds_b128 (one 16B chunk per thread).
// ---------------------------------------------------------------------------
#define BSTR 40  // LDS row stride (elems) for the 64x32 weight tile
__global__ __launch_bounds__(256) void gemm_bf16_kernel(
    const __bf16* __restrict__ A, const __bf16* __restrict__ W,
    const float* __restrict__ bias, float* __restrict__ C,
    int M, int N, int K, int relu) {
  __shared__ __bf16 sB[2][64 * BSTR];
  int tid = threadIdx.x;
  int lane = tid & 31, wave = tid >> 5;
  int m0 = blockIdx.x * 128 + wave * 16;
  int n0 = blockIdx.y * 64;

  int srow = tid >> 2;   // 0..63 : local N row
  int sch = tid & 3;     // 16B chunk within the 32-K row
  const __bf16* wrow = W + (size_t)(n0 + srow) * K + sch * 8;
  unsigned ldsrow =
      (unsigned)(uintptr_t)&sB[0][srow * BSTR + sch * 8];
  unsigned ldsbufstride = (unsigned)(64 * BSTR * sizeof(__bf16));

  // prologue: fill buffer 0 with k0 = 0
  async_copy_b128(ldsrow, wrow);
  wait_async0();
  __syncthreads();

  v8f acc[4] = {};
  int nk = K >> 5;
  for (int kk = 0; kk < nk; ++kk) {
    int cur = kk & 1;
    if (kk + 1 < nk)  // prefetch next 64x32 tile into the other buffer
      async_copy_b128(ldsrow + (cur ^ 1) * ldsbufstride, wrow + ((kk + 1) << 5));
    v16bf a = load_a_frag(A, K, m0, kk << 5, lane);
#pragma unroll
    for (int t = 0; t < 4; ++t) {
      const __bf16* bp =
          &sB[cur][(t * 16 + (lane & 15)) * BSTR + (lane >> 4) * 16];
      acc[t] = wmma_bf16(a, load16_contig(bp), acc[t]);
    }
    wait_async0();
    __syncthreads();
  }

  int rbase = m0 + (lane >> 4) * 8;
  int cl = lane & 15;
#pragma unroll
  for (int t = 0; t < 4; ++t) {
    int col = n0 + t * 16 + cl;
    float bv = bias ? bias[col] : 0.0f;
#pragma unroll
    for (int r = 0; r < 8; ++r) {
      float v = acc[t][r] + bv;
      if (relu) v = fmaxf(v, 0.0f);
      C[(size_t)(rbase + r) * N + col] = v;
    }
  }
}

// ---------------------------------------------------------------------------
// Attention with distance-decay pipeline. One block per (b, h, 16-query tile).
// Phase 1: S = scale * Q K^T via WMMA into LDS (16 x 1024 f32 = 64 KB).
// Phase 2: per-row softmax -> cumsum (wave32 shfl scan) -> decay -> softmax.
// Phase 3: O = attn @ V; the 32 x dv V slab is async-staged into LDS per
//          k-chunk (shared by all waves), B fragments read from LDS.
// q == k (kq_same with identical inputs), so Kp serves as both.
// ---------------------------------------------------------------------------
#define VSTRPAD 8
__global__ __launch_bounds__(256) void attn_kernel(
    const __bf16* __restrict__ Kp,  // (B*L, d)    head h at cols [h*dk, ..)
    const __bf16* __restrict__ Vp,  // (B*L, vlen) head h at cols [h*dv, ..)
    const float* __restrict__ gam,  // (H,)
    float* __restrict__ O,          // (B*L, vlen)
    int d, int vlen, int dk, int dv, int mask_k, int zero_pad) {
  __shared__ float sS[16][1024];               // 64 KB score slab
  __shared__ __bf16 sV[32 * (256 + VSTRPAD)];  // 32 x dv V slab (padded)
  const int L = 1024;
  int lane = threadIdx.x & 31, wave = threadIdx.x >> 5;
  int q0 = blockIdx.x * 16;
  int h = blockIdx.y;
  size_t rowbase = (size_t)blockIdx.z * L;
  const __bf16* pK = Kp + rowbase * d;
  const __bf16* pV = Vp + rowbase * vlen;
  float scale = rsqrtf((float)dk);
  int dk32 = dk >> 5;

  // ---- Phase 1: scores ----
  v16bf aq[8];
  for (int kc = 0; kc < dk32; ++kc)
    aq[kc] = load_a_frag(pK, d, q0, h * dk + kc * 32, lane);
  for (int kt = wave; kt < 64; kt += 8) {
    v8f acc = {};
    for (int kc = 0; kc < dk32; ++kc) {
      v16bf bfrag = load_b_frag(pK, d, kt * 16, h * dk + kc * 32, lane);
      acc = wmma_bf16(aq[kc], bfrag, acc);
    }
    int r0 = (lane >> 4) * 8;
    int c = kt * 16 + (lane & 15);
#pragma unroll
    for (int r = 0; r < 8; ++r) sS[r0 + r][c] = acc[r] * scale;
  }
  __syncthreads();

  // ---- Phase 2: decay pipeline (wave w owns rows w, w+8) ----
  float g = gam[h];
  float gamma = (g > 20.f) ? -g : -log1pf(expf(g));  // -softplus(gam)
  for (int rr = wave; rr < 16; rr += 8) {
    float* srow = sS[rr];
    int qi = q0 + rr;
    int lim = qi + mask_k;  // allowed: j < lim
    bool zp_row = (zero_pad && qi == 0);
    if (lim <= 0 || zp_row) {
      for (int c = 0; c < 32; ++c) srow[c * 32 + lane] = 0.0f;
      continue;
    }
    float m = -INFINITY;
    for (int c = 0; c < 32; ++c) {
      int j = c * 32 + lane;
      if (j < lim) m = fmaxf(m, srow[j]);
    }
#pragma unroll
    for (int off = 16; off > 0; off >>= 1) m = fmaxf(m, __shfl_xor(m, off, 32));
    float s = 0.f;
    for (int c = 0; c < 32; ++c) {
      int j = c * 32 + lane;
      if (j < lim) s += expf(srow[j] - m);
    }
#pragma unroll
    for (int off = 16; off > 0; off >>= 1) s += __shfl_xor(s, off, 32);
    float inv = 1.0f / s;
    float carry = 0.f, m2 = -INFINITY;
    for (int c = 0; c < 32; ++c) {
      int j = c * 32 + lane;
      bool ok = j < lim;
      float sc = srow[j];
      float e = ok ? expf(sc - m) * inv : 0.f;
      float x = e;
#pragma unroll
      for (int off = 1; off < 32; off <<= 1) {
        float t = __shfl_up(x, off, 32);
        if (lane >= off) x += t;
      }
      float cum = carry + x;
      carry += __shfl(x, 31, 32);
      float pos = fabsf((float)(qi - j));
      float dist = sqrtf(fmaxf((1.0f - cum) * pos, 0.0f));  // tot == 1
      float eff = fminf(fmaxf(expf(dist * gamma), 1e-5f), 1e5f);
      float t2 = sc * eff;
      if (ok) { srow[j] = t2; m2 = fmaxf(m2, t2); }
      else    srow[j] = 0.f;
    }
#pragma unroll
    for (int off = 16; off > 0; off >>= 1) m2 = fmaxf(m2, __shfl_xor(m2, off, 32));
    float s2 = 0.f;
    for (int c = 0; c < 32; ++c) {
      int j = c * 32 + lane;
      if (j < lim) s2 += expf(srow[j] - m2);
    }
#pragma unroll
    for (int off = 16; off > 0; off >>= 1) s2 += __shfl_xor(s2, off, 32);
    float inv2 = 1.0f / s2;
    for (int c = 0; c < 32; ++c) {
      int j = c * 32 + lane;
      srow[j] = (j < lim) ? expf(srow[j] - m2) * inv2 : 0.0f;
    }
  }
  __syncthreads();

  // ---- Phase 3: O = attn @ V (V slab async-staged into LDS per k-chunk) ---
  int vstr = dv + VSTRPAD;
  int cshift = (dv == 256) ? 5 : 4;  // log2(chunks per row = dv/8)
  int cmask = (dv >> 3) - 1;
  int nchunk = dv >> 6;              // per-thread 16B chunks (4 or 2)
  int myn = dv >> 7;                 // owned 16-col tiles per wave (2 or 1)
  v8f oacc[2] = {};
  for (int kc = 0; kc < 32; ++kc) {
    int kb = kc * 32;
    for (int c = 0; c < nchunk; ++c) {
      int t = c * 256 + threadIdx.x;
      int row = t >> cshift;
      int col8 = (t & cmask) * 8;
      unsigned lds = (unsigned)(uintptr_t)&sV[row * vstr + col8];
      async_copy_b128(lds, pV + (size_t)(kb + row) * vlen + h * dv + col8);
    }
    wait_async0();
    __syncthreads();
    // A fragment from LDS scores (f32 -> bf16)
    int arow = lane & 15;
    int kbA = kb + (lane >> 4) * 16;
    v16bf a;
#pragma unroll
    for (int i = 0; i < 8; ++i) {
      a[i]     = (__bf16)sS[arow][kbA + i];
      a[i + 8] = (__bf16)sS[arow][kbA + 8 + i];
    }
    int rb = (lane >> 4) * 16;
    for (int u = 0; u < myn; ++u) {
      int nt = wave + u * 8;
      int colL = nt * 16 + (lane & 15);
      v16bf b;
#pragma unroll
      for (int i = 0; i < 16; ++i) b[i] = sV[(rb + i) * vstr + colL];
      oacc[u] = wmma_bf16(a, b, oacc[u]);
    }
    __syncthreads();  // all reads done before next slab overwrites sV
  }
  int r0 = q0 + (lane >> 4) * 8;
  for (int u = 0; u < myn; ++u) {
    int nt = wave + u * 8;
    int col = h * dv + nt * 16 + (lane & 15);
#pragma unroll
    for (int r = 0; r < 8; ++r)
      O[(rowbase + r0 + r) * vlen + col] = oacc[u][r];
  }
}

// ---------------------------------------------------------------------------
// LayerNorm(X + R) * g + b  -> out (f32) and optional bf16 copy.
// One block (256 threads) per row; d in {1024, 2048}.
// ---------------------------------------------------------------------------
__global__ __launch_bounds__(256) void ln_kernel(
    const float* __restrict__ X, const float* __restrict__ R,
    const float* __restrict__ g, const float* __restrict__ bta,
    float* __restrict__ out, __bf16* __restrict__ obf, int d) {
  __shared__ float sred[8];
  int row = blockIdx.x;
  const float* xr = X + (size_t)row * d;
  const float* rr = R + (size_t)row * d;
  int nper = d >> 8;
  float lv[8];
  float s = 0.f;
  for (int t = 0; t < nper; ++t) {
    int i = t * 256 + threadIdx.x;
    float v = xr[i] + rr[i];
    lv[t] = v;
    s += v;
  }
  int lane = threadIdx.x & 31, wave = threadIdx.x >> 5;
#pragma unroll
  for (int off = 16; off > 0; off >>= 1) s += __shfl_xor(s, off, 32);
  if (lane == 0) sred[wave] = s;
  __syncthreads();
  float tot = 0.f;
  for (int w = 0; w < 8; ++w) tot += sred[w];
  float mu = tot / d;
  __syncthreads();
  float v2 = 0.f;
  for (int t = 0; t < nper; ++t) { float dd = lv[t] - mu; v2 += dd * dd; }
#pragma unroll
  for (int off = 16; off > 0; off >>= 1) v2 += __shfl_xor(v2, off, 32);
  if (lane == 0) sred[wave] = v2;
  __syncthreads();
  float tv = 0.f;
  for (int w = 0; w < 8; ++w) tv += sred[w];
  float rstd = rsqrtf(tv / d + 1e-5f);
  for (int t = 0; t < nper; ++t) {
    int i = t * 256 + threadIdx.x;
    float o = g[i] * (lv[t] - mu) * rstd + bta[i];
    out[(size_t)row * d + i] = o;
    if (obf) obf[(size_t)row * d + i] = (__bf16)o;
  }
}

// ---------------------------------------------------------------------------
// Host launcher
// ---------------------------------------------------------------------------
extern "C" void kernel_launch(void* const* d_in, const int* in_sizes, int n_in,
                              void* d_out, int out_size, void* d_ws,
                              size_t ws_size, hipStream_t stream) {
  (void)in_sizes; (void)n_in; (void)out_size; (void)ws_size;
  const int Bb = 8, Ll = 1024, Hh = 8;
  const int BL = Bb * Ll;

  const float* q_embed = (const float*)d_in[0];
  const float* qa_embed = (const float*)d_in[1];
  int idx = 2;
  auto nf = [&]() { return (const float*)d_in[idx++]; };
  // y block (d = 2048, vlen = 2048, with FFN)
  const float *Wk_y=nf(),*bk_y=nf(),*Wv_y=nf(),*bv_y=nf(),*Wo_y=nf(),*bo_y=nf(),
              *gam_y=nf(),*ln1g_y=nf(),*ln1b_y=nf(),*W1_y=nf(),*b1_y=nf(),
              *W2_y=nf(),*b2_y=nf(),*ln2g_y=nf(),*ln2b_y=nf();
  // x block (d = 1024, vlen = 1024, no FFN)
  const float *Wk_x=nf(),*bk_x=nf(),*Wv_x=nf(),*bv_x=nf(),*Wo_x=nf(),*bo_x=nf(),
              *gam_x=nf(),*ln1g_x=nf(),*ln1b_x=nf();
  // h block (q/k d = 1024, vlen = 2048, with FFN)
  const float *Wk_h=nf(),*bk_h=nf(),*Wv_h=nf(),*bv_h=nf(),*Wo_h=nf(),*bo_h=nf(),
              *gam_h=nf(),*ln1g_h=nf(),*ln1b_h=nf(),*W1_h=nf(),*b1_h=nf(),
              *W2_h=nf(),*b2_h=nf(),*ln2g_h=nf(),*ln2b_h=nf();

  // workspace bump allocator (256B aligned)
  char* wp = (char*)d_ws;
  auto af = [&](size_t e) { float* p = (float*)wp; wp += ((e * 4) + 255) & ~(size_t)255; return p; };
  auto ab = [&](size_t e) { __bf16* p = (__bf16*)wp; wp += ((e * 2) + 255) & ~(size_t)255; return p; };

  float* f0 = af((size_t)BL * 4096);
  float* f1 = af((size_t)BL * 2048);
  float* f2 = af((size_t)BL * 2048);
  float* f3 = af((size_t)BL * 2048);
  float* f4 = af((size_t)BL * 1024);
  float* f5 = af((size_t)BL * 1024);
  __bf16* bb0 = ab((size_t)BL * 4096);
  __bf16* bb1 = ab((size_t)BL * 2048);
  __bf16* bb2 = ab((size_t)BL * 2048);
  __bf16* bb3 = ab((size_t)BL * 2048);
  __bf16* bb4 = ab((size_t)BL * 2048);
  __bf16 *wWk_y = ab((size_t)2048 * 2048), *wWv_y = ab((size_t)2048 * 2048),
         *wWo_y = ab((size_t)2048 * 2048), *wW1_y = ab((size_t)4096 * 2048),
         *wW2_y = ab((size_t)2048 * 4096);
  __bf16 *wWk_x = ab((size_t)1024 * 1024), *wWv_x = ab((size_t)1024 * 1024),
         *wWo_x = ab((size_t)1024 * 1024);
  __bf16 *wWk_h = ab((size_t)1024 * 1024), *wWv_h = ab((size_t)2048 * 2048),
         *wWo_h = ab((size_t)1024 * 2048), *wW1_h = ab((size_t)4096 * 1024),
         *wW2_h = ab((size_t)1024 * 4096);

  auto CVT = [&](const float* s, __bf16* dst, size_t n) {
    unsigned blocks = (unsigned)((n + 1023) / 1024);
    cvt_bf16_kernel<<<blocks, 256, 0, stream>>>(s, dst, n);
  };
  auto GEMM = [&](const __bf16* A, const __bf16* W, const float* bias, float* C,
                  int M, int N, int K, int relu) {
    gemm_bf16_kernel<<<dim3(M / 128, N / 64), 256, 0, stream>>>(A, W, bias, C, M, N, K, relu);
  };
  auto ATTN = [&](const __bf16* Kp, const __bf16* Vp, const float* gm, float* O,
                  int dd, int vl, int mk, int zp) {
    attn_kernel<<<dim3(Ll / 16, Hh, Bb), 256, 0, stream>>>(
        Kp, Vp, gm, O, dd, vl, dd / Hh, vl / Hh, mk, zp);
  };
  auto LNK = [&](const float* X, const float* R, const float* g,
                 const float* bta, float* out, __bf16* obf, int dd) {
    ln_kernel<<<BL, 256, 0, stream>>>(X, R, g, bta, out, obf, dd);
  };

  // ---- weights -> bf16 ----
  CVT(Wk_y, wWk_y, (size_t)2048 * 2048); CVT(Wv_y, wWv_y, (size_t)2048 * 2048);
  CVT(Wo_y, wWo_y, (size_t)2048 * 2048); CVT(W1_y, wW1_y, (size_t)4096 * 2048);
  CVT(W2_y, wW2_y, (size_t)2048 * 4096);
  CVT(Wk_x, wWk_x, (size_t)1024 * 1024); CVT(Wv_x, wWv_x, (size_t)1024 * 1024);
  CVT(Wo_x, wWo_x, (size_t)1024 * 1024);
  CVT(Wk_h, wWk_h, (size_t)1024 * 1024); CVT(Wv_h, wWv_h, (size_t)2048 * 2048);
  CVT(Wo_h, wWo_h, (size_t)1024 * 2048); CVT(W1_h, wW1_h, (size_t)4096 * 1024);
  CVT(W2_h, wW2_h, (size_t)1024 * 4096);

  // ================= block y (self-attn on qa, width 2048, FFN) ============
  CVT(qa_embed, bb1, (size_t)BL * 2048);
  GEMM(bb1, wWk_y, bk_y, f1, BL, 2048, 2048, 0);  // K (== Q, kq_same)
  CVT(f1, bb2, (size_t)BL * 2048);
  GEMM(bb1, wWv_y, bv_y, f2, BL, 2048, 2048, 0);  // V
  CVT(f2, bb3, (size_t)BL * 2048);
  ATTN(bb2, bb3, gam_y, f2, 2048, 2048, /*mask_k=*/1, /*zero_pad=*/0);
  CVT(f2, bb4, (size_t)BL * 2048);
  GEMM(bb4, wWo_y, bo_y, f1, BL, 2048, 2048, 0);  // output proj
  LNK(f1, qa_embed, ln1g_y, ln1b_y, f3, bb1, 2048);
  GEMM(bb1, wW1_y, b1_y, f0, BL, 4096, 2048, 1);  // FFN1 + ReLU
  CVT(f0, bb0, (size_t)BL * 4096);
  GEMM(bb0, wW2_y, b2_y, f1, BL, 2048, 4096, 0);  // FFN2
  LNK(f1, f3, ln2g_y, ln2b_y, f3, bb3, 2048);     // y -> f3 (f32), bb3 (bf16)

  // ================= block x (self-attn on q, width 1024, no FFN) ==========
  CVT(q_embed, bb1, (size_t)BL * 1024);
  GEMM(bb1, wWk_x, bk_x, f1, BL, 1024, 1024, 0);
  CVT(f1, bb2, (size_t)BL * 1024);
  GEMM(bb1, wWv_x, bv_x, f2, BL, 1024, 1024, 0);
  CVT(f2, bb4, (size_t)BL * 1024);
  ATTN(bb2, bb4, gam_x, f2, 1024, 1024, /*mask_k=*/1, /*zero_pad=*/0);
  CVT(f2, bb2, (size_t)BL * 1024);
  GEMM(bb2, wWo_x, bo_x, f1, BL, 1024, 1024, 0);
  LNK(f1, q_embed, ln1g_x, ln1b_x, f4, bb1, 1024);  // x -> f4, bb1

  // ================= block h (cross: q/k = x, values = y, FFN) =============
  GEMM(bb1, wWk_h, bk_h, f1, BL, 1024, 1024, 0);    // K from x
  CVT(f1, bb2, (size_t)BL * 1024);
  GEMM(bb3, wWv_h, bv_h, f2, BL, 2048, 2048, 0);    // V from y
  CVT(f2, bb0, (size_t)BL * 2048);
  ATTN(bb2, bb0, gam_h, f2, 1024, 2048, /*mask_k=*/0, /*zero_pad=*/1);
  CVT(f2, bb4, (size_t)BL * 2048);
  GEMM(bb4, wWo_h, bo_h, f1, BL, 1024, 2048, 0);
  LNK(f1, f4, ln1g_h, ln1b_h, f5, bb1, 1024);
  GEMM(bb1, wW1_h, b1_h, f0, BL, 4096, 1024, 1);    // FFN1 + ReLU
  CVT(f0, bb0, (size_t)BL * 4096);
  GEMM(bb0, wW2_h, b2_h, f1, BL, 1024, 4096, 0);    // FFN2
  LNK(f1, f5, ln2g_h, ln2b_h, (float*)d_out, nullptr, 1024);  // final output
}
